// Model_78649441124436
// MI455X (gfx1250) — matrix operand order repaired and checked by reading, not compiled, not run on hardware
//
#include <hip/hip_runtime.h>
#include <hip/hip_bf16.h>

typedef __attribute__((ext_vector_type(16))) _Float16 v16h;
typedef __attribute__((ext_vector_type(8)))  float    v8f;

// ---------------- degree / normalization ----------------

__global__ void deg_init_kernel(int* __restrict__ deg, int N) {
  int i = blockIdx.x * blockDim.x + threadIdx.x;
  if (i < N) deg[i] = 1;                 // self loop: deg = count(dst) + 1
}

__global__ void deg_count_kernel(const int* __restrict__ ei, int* __restrict__ deg, int E) {
  int e = blockIdx.x * blockDim.x + threadIdx.x;
  if (e < E) atomicAdd(&deg[ei[E + e]], 1);   // dst = edge_index[1][e]
}

__global__ void deg_rsqrt_kernel(float* __restrict__ dis, int N) {
  int i = blockIdx.x * blockDim.x + threadIdx.x;
  if (i < N) {
    int d = ((const int*)dis)[i];
    dis[i] = rsqrtf((float)d);
  }
}

// ---------------- one-hot embed: z[n] = [W1[type]+b1 , W2[pos]+b2] ----------------

__global__ void embed_kernel(const int* __restrict__ types, const int* __restrict__ pos,
                             const float* __restrict__ W1, const float* __restrict__ b1,
                             const float* __restrict__ W2, const float* __restrict__ b2,
                             float* __restrict__ z, int N) {
  int t = blockIdx.x * blockDim.x + threadIdx.x;
  if (t >= N * 16) return;
  int n = t >> 4, k = t & 15;
  z[n * 32 + k]      = W1[types[n] * 16 + k] + b1[k];
  z[n * 32 + 16 + k] = W2[pos[n]   * 16 + k] + b2[k];
}

// ---------------- dense part: h = relu?(X) @ W  (WMMA f32<-f16 16x16x32) ----------------
// One wave per 16-row tile; fuses input ReLU and the self-loop term agg = h*dis^2 + bias.
// Layouts per CDNA5 ISA 7.12.2:
//   A (16-bit, 16x32): elem j of lane -> M=lane&15, K = (j<8 ? j : 8+j) + 8*(lane>>4)
//     => per lane: two contiguous 8-float runs of its row -> float4 loads
//   B (16-bit, 32x16): elem j of lane -> N=lane&15, K = (j<8 ? j : 8+j) + 8*(lane>>4)
//   C/D (32-bit 16x16): c[i] of lane  -> M = i + 8*(lane>>4), N = lane&15

template<int KIN, bool RELU_IN>
__global__ void gemm_wmma_kernel(const float* __restrict__ X,
                                 const float* __restrict__ W,
                                 const float* __restrict__ bias,
                                 const float* __restrict__ dis,
                                 float* __restrict__ h,
                                 float* __restrict__ agg,
                                 int N, int ntiles) {
  int wave = threadIdx.x >> 5;
  int lane = threadIdx.x & 31;
  int tile = blockIdx.x * (blockDim.x >> 5) + wave;
  if (tile >= ntiles) return;            // wave-uniform: EXEC stays all-1s for WMMA
  int half = lane >> 4;
  int nn   = lane & 15;
  int m0   = tile << 4;

  v16h a, b;
  v8f  c = {};

  // B (weights): 16 strided scalar loads, W is tiny and L1-hot
#pragma unroll
  for (int j = 0; j < 16; ++j) {
    int K = (j < 8 ? j : 8 + j) + 8 * half;
    b[j] = (K < KIN) ? (_Float16)W[K * 16 + nn] : (_Float16)0.0f;
  }

  // A (activations): vectorized b128 loads of the two contiguous 8-float runs
  int mrow = m0 + nn; if (mrow >= N) mrow = N - 1;   // clamp for tail tiles
  const float4* xv = (const float4*)(X + (size_t)mrow * KIN);
  float buf[16];
  {
    float4 p0 = xv[2 * half];          // K = 8*half + 0..3
    float4 p1 = xv[2 * half + 1];      // K = 8*half + 4..7
    buf[0] = p0.x; buf[1] = p0.y; buf[2] = p0.z; buf[3] = p0.w;
    buf[4] = p1.x; buf[5] = p1.y; buf[6] = p1.z; buf[7] = p1.w;
    if (KIN == 32) {
      float4 p2 = xv[4 + 2 * half];    // K = 16 + 8*half + 0..3
      float4 p3 = xv[5 + 2 * half];    // K = 16 + 8*half + 4..7
      buf[8]  = p2.x; buf[9]  = p2.y; buf[10] = p2.z; buf[11] = p2.w;
      buf[12] = p3.x; buf[13] = p3.y; buf[14] = p3.z; buf[15] = p3.w;
    } else {
#pragma unroll
      for (int j = 8; j < 16; ++j) buf[j] = 0.0f;
    }
  }
#pragma unroll
  for (int j = 0; j < 16; ++j) {
    float v = buf[j];
    if (RELU_IN) v = fmaxf(v, 0.0f);
    a[j] = (_Float16)v;
  }

  c = __builtin_amdgcn_wmma_f32_16x16x32_f16(
      /*neg_a=*/false, a, /*neg_b=*/false, b,
      /*c_mod=*/(short)0, c, /*reuse_a=*/false, /*reuse_b=*/false);

  float bv = bias[nn];
  if (m0 + 16 <= N) {                   // fast path: full tile, no per-row guards
#pragma unroll
    for (int i = 0; i < 8; ++i) {
      int r = m0 + i + 8 * half;
      float ds = dis[r];
      float v  = c[i];
      h[r * 16 + nn]   = v;
      agg[r * 16 + nn] = v * ds * ds + bv;
    }
  } else {
#pragma unroll
    for (int i = 0; i < 8; ++i) {
      int r = m0 + i + 8 * half;
      if (r < N) {
        float ds = dis[r];
        float v  = c[i];
        h[r * 16 + nn]   = v;
        agg[r * 16 + nn] = v * ds * ds + bv;
      }
    }
  }
}

// ---------------- sparse part: agg[dst] += h[src] * dis[src]*dis[dst] ----------------
// 16 consecutive lanes per edge -> 64B coalesced gather + 64B coalesced atomics (L2-resident).

__global__ void edge_scatter_kernel(const int* __restrict__ ei, const float* __restrict__ dis,
                                    const float* __restrict__ h, float* __restrict__ agg, int E) {
  int t = blockIdx.x * blockDim.x + threadIdx.x;
  if (t >= E * 16) return;
  int e = t >> 4;
  int k = t & 15;
  int s = ei[e];
  int d = ei[E + e];
  float coef = dis[s] * dis[d];
  atomicAdd(&agg[(size_t)d * 16 + k], h[(size_t)s * 16 + k] * coef);
}

__global__ void zero_kernel(float* __restrict__ p, int n) {
  int i = blockIdx.x * blockDim.x + threadIdx.x;
  if (i < n) p[i] = 0.0f;
}

// ---------------- global mean pool + output head ----------------

__global__ void pool_sum_kernel(const float* __restrict__ x, const int* __restrict__ batch,
                                float* __restrict__ pooled, int N) {
  int t = blockIdx.x * blockDim.x + threadIdx.x;
  if (t >= N * 16) return;
  int n = t >> 4, k = t & 15;
  atomicAdd(&pooled[(size_t)batch[n] * 16 + k], x[t]);
}

__global__ void pool_cnt_kernel(const int* __restrict__ batch, float* __restrict__ cnt, int N) {
  int i = blockIdx.x * blockDim.x + threadIdx.x;
  if (i < N) atomicAdd(&cnt[batch[i]], 1.0f);
}

__global__ void head_kernel(const float* __restrict__ pooled, const float* __restrict__ cnt,
                            const float* __restrict__ Wo, const float* __restrict__ bo,
                            float* __restrict__ out, int G) {
  int g = blockIdx.x * blockDim.x + threadIdx.x;
  if (g >= G) return;
  float s = 0.0f;
#pragma unroll
  for (int k = 0; k < 16; ++k) s += pooled[g * 16 + k] * Wo[k];
  out[g] = s / fmaxf(cnt[g], 1.0f) + bo[0];
}

// ---------------- launcher ----------------

extern "C" void kernel_launch(void* const* d_in, const int* in_sizes, int n_in,
                              void* d_out, int out_size, void* d_ws, size_t ws_size,
                              hipStream_t stream) {
  const int*   types = (const int*)d_in[0];
  const int*   pos   = (const int*)d_in[1];
  const int*   ei    = (const int*)d_in[2];
  const int*   batch = (const int*)d_in[3];
  const float* W1    = (const float*)d_in[4];
  const float* b1    = (const float*)d_in[5];
  const float* W2    = (const float*)d_in[6];
  const float* b2    = (const float*)d_in[7];
  const float* Wg0   = (const float*)d_in[8];
  const float* bg0   = (const float*)d_in[9];
  const float* Wg1   = (const float*)d_in[10];
  const float* bg1   = (const float*)d_in[11];
  const float* Wg2   = (const float*)d_in[12];
  const float* bg2   = (const float*)d_in[13];
  const float* Wo    = (const float*)d_in[14];
  const float* bo    = (const float*)d_in[15];
  float* out = (float*)d_out;

  const int N = in_sizes[0];
  const int E = in_sizes[2] / 2;
  const int G = out_size;

  // workspace layout (floats): dis | z(32N) | h(16N) | aggA(16N) | aggB(16N) | pooled(16G) | cnt(G)
  float* F      = (float*)d_ws;
  float* dis    = F;
  float* z      = dis + N;
  float* h      = z + (size_t)32 * N;
  float* aggA   = h + (size_t)16 * N;
  float* aggB   = aggA + (size_t)16 * N;
  float* pooled = aggB + (size_t)16 * N;
  float* cnt    = pooled + (size_t)16 * G;   // zeroed together with pooled

  auto cdiv = [](long long a, long long b) { return (int)((a + b - 1) / b); };

  const int ntiles       = cdiv(N, 16);
  const int gemm_blocks  = cdiv(ntiles, 8);                 // 8 waves / 256-thread block
  const int edge_blocks  = cdiv((long long)E * 16, 256);
  const int nodef_blocks = cdiv((long long)N * 16, 256);

  // GCN symmetric normalization
  deg_init_kernel <<<cdiv(N, 256), 256, 0, stream>>>((int*)dis, N);
  deg_count_kernel<<<cdiv(E, 256), 256, 0, stream>>>(ei, (int*)dis, E);
  deg_rsqrt_kernel<<<cdiv(N, 256), 256, 0, stream>>>(dis, N);

  // embedding
  embed_kernel<<<nodef_blocks, 256, 0, stream>>>(types, pos, W1, b1, W2, b2, z, N);

  // layer 0 (K=32): z -> h, aggA
  gemm_wmma_kernel<32, false><<<gemm_blocks, 256, 0, stream>>>(z, Wg0, bg0, dis, h, aggA, N, ntiles);
  edge_scatter_kernel<<<edge_blocks, 256, 0, stream>>>(ei, dis, h, aggA, E);

  // layer 1 (K=16): relu(aggA) -> h, aggB   (ReLU fused into A-load)
  gemm_wmma_kernel<16, true><<<gemm_blocks, 256, 0, stream>>>(aggA, Wg1, bg1, dis, h, aggB, N, ntiles);
  edge_scatter_kernel<<<edge_blocks, 256, 0, stream>>>(ei, dis, h, aggB, E);

  // layer 2 (K=16): relu(aggB) -> h, aggA; no output activation
  gemm_wmma_kernel<16, true><<<gemm_blocks, 256, 0, stream>>>(aggB, Wg2, bg2, dis, h, aggA, N, ntiles);
  edge_scatter_kernel<<<edge_blocks, 256, 0, stream>>>(ei, dis, h, aggA, E);

  // mean pool + head (pooled 16G and cnt G are contiguous -> zero 17G)
  zero_kernel<<<cdiv((long long)17 * G, 256), 256, 0, stream>>>(pooled, 17 * G);
  pool_sum_kernel<<<nodef_blocks, 256, 0, stream>>>(aggA, batch, pooled, N);
  pool_cnt_kernel<<<cdiv(N, 256), 256, 0, stream>>>(batch, cnt, N);
  head_kernel<<<cdiv(G, 256), 256, 0, stream>>>(pooled, cnt, Wo, bo, out, G);
}
